// SubtourDetector_44487271252594
// MI455X (gfx1250) — matrix operand order repaired and checked by reading, not compiled, not run on hardware
//
#include <hip/hip_runtime.h>
#include <hip/hip_bf16.h>

// ---------------- problem constants (match reference) ----------------
#define NN      32768       // total nodes (B*NP)
#define NPG     512         // nodes per graph
#define NB      64          // graphs
#define NE      1048576     // edges
#define FIN     16
#define HD      128         // hidden
#define KA      256         // A-matrix K (2*HD)
#define BN_EPS  1e-5f

typedef _Float16 v16h __attribute__((ext_vector_type(16)));
typedef _Float16 v8h  __attribute__((ext_vector_type(8)));
typedef float    v8f  __attribute__((ext_vector_type(8)));

static __device__ __forceinline__ void atomAddAgent(float* p, float v) {
    __hip_atomic_fetch_add(p, v, __ATOMIC_RELAXED, __HIP_MEMORY_SCOPE_AGENT);
}
static __device__ __forceinline__ void atomAddWG(float* p, float v) {
    __hip_atomic_fetch_add(p, v, __ATOMIC_RELAXED, __HIP_MEMORY_SCOPE_WORKGROUP);
}

// ---------------- zero fill ----------------
__global__ void k_zero(float* p, int n) {
    int i = blockIdx.x * 256 + threadIdx.x;
    if (i < n) p[i] = 0.0f;
}

// ---------------- x0 = nf @ W_pre + b_pre  (K=16, scalar f32) ----------------
__global__ __launch_bounds__(256) void k_pre(const float* __restrict__ nf,
                                             const float* __restrict__ Wp,
                                             const float* __restrict__ bp,
                                             float* __restrict__ x0) {
    int idx = blockIdx.x * 256 + threadIdx.x;     // grid = NN*HD/256
    int n = idx >> 7, h = idx & 127;
    float acc = bp[h];
    const float* row = nf + n * FIN;
#pragma unroll
    for (int f = 0; f < FIN; ++f) acc = fmaf(row[f], Wp[f * HD + h], acc);
    x0[idx] = acc;
}

// ---------------- Wt[o][k] = f16( k<128 ? Wrel[k][o] : Wroot[k-128][o] ) ----------------
__global__ __launch_bounds__(256) void k_prep_wt(const float* __restrict__ Wrel,
                                                 const float* __restrict__ Wroot,
                                                 _Float16* __restrict__ Wt) {
    int o = blockIdx.x;            // 128 blocks
    int k = threadIdx.x;           // 256 threads
    float v = (k < HD) ? Wrel[k * HD + o] : Wroot[(k - HD) * HD + o];
    Wt[o * KA + k] = (_Float16)v;
}

// ---------------- scatter: agg[dst] += w * trans(x[src]); one wave per edge ----------------
__global__ __launch_bounds__(256) void k_scatter(const int* __restrict__ src,
                                                 const int* __restrict__ dst,
                                                 const float* __restrict__ ew,
                                                 const float* __restrict__ xin,
                                                 const float* __restrict__ scale,
                                                 const float* __restrict__ shift,
                                                 float* __restrict__ agg) {
    int tid = threadIdx.x;
    int e = blockIdx.x * 8 + (tid >> 5);          // grid = NE/8 exactly
    int lane = tid & 31;
    int s = src[e], d = dst[e];
    float w = ew[e];
    int h = lane * 4;
    float4 v = *(const float4*)(xin + (size_t)s * HD + h);
    if (scale) {   // BN + ReLU applied on the fly (layer 2 input)
        v.x = fmaxf(fmaf(v.x, scale[h + 0], shift[h + 0]), 0.0f);
        v.y = fmaxf(fmaf(v.y, scale[h + 1], shift[h + 1]), 0.0f);
        v.z = fmaxf(fmaf(v.z, scale[h + 2], shift[h + 2]), 0.0f);
        v.w = fmaxf(fmaf(v.w, scale[h + 3], shift[h + 3]), 0.0f);
    }
    float* ap = agg + (size_t)d * HD + h;
    atomAddAgent(ap + 0, w * v.x);
    atomAddAgent(ap + 1, w * v.y);
    atomAddAgent(ap + 2, w * v.z);
    atomAddAgent(ap + 3, w * v.w);
}

// ---------------- pack Ah = f16([agg | trans(x)])  [NN x 256] ----------------
__global__ __launch_bounds__(256) void k_pack(const float* __restrict__ agg,
                                              const float* __restrict__ xin,
                                              const float* __restrict__ scale,
                                              const float* __restrict__ shift,
                                              _Float16* __restrict__ Ah) {
    int idx = blockIdx.x * 256 + threadIdx.x;     // grid = NN*HD/256
    int n = idx >> 7, h = idx & 127;
    Ah[(size_t)n * KA + h] = (_Float16)agg[idx];
    float v = xin[idx];
    if (scale) v = fmaxf(fmaf(v, scale[h], shift[h]), 0.0f);
    Ah[(size_t)n * KA + HD + h] = (_Float16)v;
}

// ---------------- WMMA GEMM: Y[NN x 128] = Ah[NN x 256] * B + bias ----------------
// B[k][n] = Wt[n][k] (Wt is B-transposed, row-major [128][256] f16).
// One wave computes a 16x16 tile; block = 8 waves covering all 8 n-tiles of one m-tile.
__global__ __launch_bounds__(256) void k_gemm(const _Float16* __restrict__ Ah,
                                              const _Float16* __restrict__ Wt,
                                              const float* __restrict__ bias,
                                              float* __restrict__ Y) {
    int wave = threadIdx.x >> 5;
    int lane = threadIdx.x & 31;
    int mtile = blockIdx.x;                       // grid = NN/16 = 2048
    int ntile = wave;                             // 8 n-tiles
    int lhalf = lane >> 4;                        // lane group 0/1
    int l15 = lane & 15;
    int m = mtile * 16 + l15;                     // A row held by this lane
    int n = ntile * 16 + l15;                     // B/C column held by this lane
    const _Float16* arow = Ah + (size_t)m * KA;
    const _Float16* brow = Wt + (size_t)n * KA + lhalf * 16;  // B: contiguous 16-K chunk
    int kbA = lhalf * 16;                         // A: contiguous 16-K chunk
    v8f acc = {};
#pragma unroll
    for (int kc = 0; kc < 8; ++kc) {
        const _Float16* pa = arow + kc * 32;
        v8h lo = *(const v8h*)(pa + kbA);         // K = 32kc + kbA .. +7
        v8h hi = *(const v8h*)(pa + kbA + 8);     // K = 32kc + kbA + 8 .. +15
        v16h a;
#pragma unroll
        for (int i = 0; i < 8; ++i) { a[i] = lo[i]; a[8 + i] = hi[i]; }
        v16h b = *(const v16h*)(brow + kc * 32);  // K = 32kc + 16*lhalf .. +15
        acc = __builtin_amdgcn_wmma_f32_16x16x32_f16(false, a, false, b,
                                                     (short)0, acc, false, false);
    }
    float bv = bias[n];
    int mbase = mtile * 16 + lhalf * 8;           // C layout: VGPR r -> row mbase+r
#pragma unroll
    for (int r = 0; r < 8; ++r)
        Y[(size_t)(mbase + r) * HD + n] = acc[r] + bv;
}

// ---------------- per-channel sum / sumsq over all nodes ----------------
__global__ __launch_bounds__(256) void k_stats(const float* __restrict__ Y,
                                               float* __restrict__ gsum,
                                               float* __restrict__ gsq) {
    int t = threadIdx.x;
    int c = t & 127, g = t >> 7;
    float s = 0.0f, q = 0.0f;
    int r0 = blockIdx.x * 256 + g * 128;          // grid = 128 blocks -> 256 rows each
    for (int j = 0; j < 128; ++j) {
        float v = Y[(size_t)(r0 + j) * HD + c];
        s += v; q += v * v;
    }
    __shared__ float sh[256], shq[256];
    sh[t] = s; shq[t] = q;
    __syncthreads();
    if (t < 128) {
        atomAddAgent(&gsum[t], sh[t] + sh[t + 128]);
        atomAddAgent(&gsq[t],  shq[t] + shq[t + 128]);
    }
}

// ---------------- fold BN into per-channel scale/shift ----------------
__global__ void k_bn_finalize(const float* __restrict__ gsum, const float* __restrict__ gsq,
                              const float* __restrict__ gam, const float* __restrict__ bet,
                              float* __restrict__ scale, float* __restrict__ shift) {
    int c = threadIdx.x;                          // 1 block x 128
    float mu = gsum[c] * (1.0f / (float)NN);
    float var = gsq[c] * (1.0f / (float)NN) - mu * mu;   // biased var
    float sc = gam[c] * rsqrtf(var + BN_EPS);
    scale[c] = sc;
    shift[c] = bet[c] - mu * sc;
}

// ---------------- per-node softmax pooling assignment s[n][2] + ss bins ----------------
__global__ __launch_bounds__(128) void k_pool(const float* __restrict__ Y2,
                                              const float* __restrict__ scale,
                                              const float* __restrict__ shift,
                                              const float* __restrict__ Wpool,
                                              const float* __restrict__ bpool,
                                              float* __restrict__ S,
                                              float* __restrict__ ss0,
                                              float* __restrict__ ss1,
                                              float* __restrict__ ss2) {
    int n = blockIdx.x;                           // grid = NN
    int h = threadIdx.x;
    float v = fmaxf(fmaf(Y2[(size_t)n * HD + h], scale[h], shift[h]), 0.0f);
    __shared__ float r0[128], r1[128];
    r0[h] = v * Wpool[h * 2 + 0];
    r1[h] = v * Wpool[h * 2 + 1];
    __syncthreads();
    for (int s = 64; s > 0; s >>= 1) {
        if (h < s) { r0[h] += r0[h + s]; r1[h] += r1[h + s]; }
        __syncthreads();
    }
    if (h == 0) {
        float l0 = r0[0] + bpool[0], l1 = r1[0] + bpool[1];
        float mx = fmaxf(l0, l1);
        float e0 = expf(l0 - mx), e1 = expf(l1 - mx);
        float inv = 1.0f / (e0 + e1);
        float s0 = e0 * inv, s1 = e1 * inv;
        S[n * 2 + 0] = s0;
        S[n * 2 + 1] = s1;
        int b = n >> 9;
        atomAddAgent(&ss0[b], s0 * s0);
        atomAddAgent(&ss1[b], s0 * s1);
        atomAddAgent(&ss2[b], s1 * s1);
    }
}

// ---------------- out[b,k,h] = sum_n s[n,k] * x2[n,h] ----------------
__global__ __launch_bounds__(256) void k_out(const float* __restrict__ S,
                                             const float* __restrict__ Y2,
                                             const float* __restrict__ scale,
                                             const float* __restrict__ shift,
                                             float* __restrict__ outp) {
    int b = blockIdx.x;                           // grid = NB
    int t = threadIdx.x;
    int k = t >> 7, h = t & 127;
    float sc = scale[h], sf = shift[h];
    float acc = 0.0f;
    int base = b * NPG;
    for (int j = 0; j < NPG; ++j) {
        int n = base + j;
        float x = fmaxf(fmaf(Y2[(size_t)n * HD + h], sc, sf), 0.0f);
        acc = fmaf(S[n * 2 + k], x, acc);
    }
    outp[b * 256 + k * HD + h] = acc;
}

// ---------------- per-edge mincut num/den with LDS graph bins ----------------
__global__ __launch_bounds__(256) void k_edge_loss(const int* __restrict__ src,
                                                   const int* __restrict__ dst,
                                                   const float* __restrict__ ew,
                                                   const float* __restrict__ S,
                                                   float* __restrict__ num,
                                                   float* __restrict__ den) {
    __shared__ float nb[NB], db[NB];
    int t = threadIdx.x;
    if (t < NB) { nb[t] = 0.0f; db[t] = 0.0f; }
    __syncthreads();
    int e = blockIdx.x * 256 + t;                 // grid = NE/256 exactly
    int s = src[e], d = dst[e];
    float w = ew[e];
    int b = s >> 9;
    float a0 = S[s * 2], a1 = S[s * 2 + 1];
    float c0 = S[d * 2], c1 = S[d * 2 + 1];
    atomAddWG(&nb[b], w * (a0 * c0 + a1 * c1));
    atomAddWG(&db[b], w * (a0 * a0 + a1 * a1));   // den: d[n]=sum_{src=n} w folded in
    __syncthreads();
    if (t < NB) {
        atomAddAgent(&num[t], nb[t]);
        atomAddAgent(&den[t], db[t]);
    }
}

// ---------------- logits + mc_loss + o_loss ----------------
__global__ __launch_bounds__(256) void k_final(const float* __restrict__ outp,
                                               const float* __restrict__ Wpost,
                                               const float* __restrict__ bpost,
                                               const float* __restrict__ num,
                                               const float* __restrict__ den,
                                               const float* __restrict__ ss0,
                                               const float* __restrict__ ss1,
                                               const float* __restrict__ ss2,
                                               float* __restrict__ dout) {
    __shared__ float r0[256], r1[256];
    int t = threadIdx.x;
    for (int b = 0; b < NB; ++b) {
        float hv = fmaxf(outp[b * 256 + t], 0.0f);
        r0[t] = hv * Wpost[t * 2 + 0];
        r1[t] = hv * Wpost[t * 2 + 1];
        __syncthreads();
        for (int s = 128; s > 0; s >>= 1) {
            if (t < s) { r0[t] += r0[t + s]; r1[t] += r1[t + s]; }
            __syncthreads();
        }
        if (t == 0) {
            dout[b * 2 + 0] = r0[0] + bpost[0];
            dout[b * 2 + 1] = r1[0] + bpost[1];
        }
        __syncthreads();
    }
    // losses
    if (t < NB) {
        float mc = -num[t] / den[t];
        float s00 = ss0[t], s01 = ss1[t], s11 = ss2[t];
        float nrm = sqrtf(s00 * s00 + 2.0f * s01 * s01 + s11 * s11);
        float inv = 1.0f / nrm;
        const float id = 0.70710678118654752f;    // 1/sqrt(K), K=2
        float da = s00 * inv - id, dbv = s01 * inv, dc = s11 * inv - id;
        r0[t] = mc;
        r1[t] = sqrtf(da * da + 2.0f * dbv * dbv + dc * dc);
    }
    __syncthreads();
    for (int s = 32; s > 0; s >>= 1) {
        if (t < s) { r0[t] += r0[t + s]; r1[t] += r1[t + s]; }
        __syncthreads();
    }
    if (t == 0) {
        dout[128] = r0[0] * (1.0f / (float)NB);
        dout[129] = r1[0] * (1.0f / (float)NB);
    }
}

// =====================================================================
extern "C" void kernel_launch(void* const* d_in, const int* in_sizes, int n_in,
                              void* d_out, int out_size, void* d_ws, size_t ws_size,
                              hipStream_t stream) {
    // inputs (setup_inputs dict order)
    const float* node_feature = (const float*)d_in[0];
    const int*   edge_index   = (const int*)d_in[1];
    const float* edge_feature = (const float*)d_in[2];
    // d_in[3] = batch (unused: graphs are equal-sized & sorted -> b = n >> 9)
    const float* W_pre  = (const float*)d_in[4];
    const float* b_pre  = (const float*)d_in[5];
    const float* W1_rel = (const float*)d_in[6];
    const float* b1_rel = (const float*)d_in[7];
    const float* W1_root= (const float*)d_in[8];
    const float* g1     = (const float*)d_in[9];
    const float* be1    = (const float*)d_in[10];
    const float* W2_rel = (const float*)d_in[11];
    const float* b2_rel = (const float*)d_in[12];
    const float* W2_root= (const float*)d_in[13];
    const float* g2     = (const float*)d_in[14];
    const float* be2    = (const float*)d_in[15];
    const float* W_pool = (const float*)d_in[16];
    const float* b_pool = (const float*)d_in[17];
    const float* W_post = (const float*)d_in[18];
    const float* b_post = (const float*)d_in[19];

    const int* src = edge_index;
    const int* dst = edge_index + NE;
    float* dout = (float*)d_out;

    // ---- workspace carve (≈64.5 MB) ----
    float* buf0 = (float*)d_ws;                   // x0, then agg2
    float* buf1 = buf0 + (size_t)NN * HD;         // agg1, then y1
    float* buf2 = buf1 + (size_t)NN * HD;         // y2
    _Float16* Ah  = (_Float16*)(buf2 + (size_t)NN * HD);  // [NN x 256] f16
    _Float16* Wt1 = Ah + (size_t)NN * KA;
    _Float16* Wt2 = Wt1 + HD * KA;
    float* small = (float*)(Wt2 + HD * KA);
    float* d_sum  = small;            // 128
    float* d_sq   = small + 128;      // 128
    float* scale1 = small + 256;      // 128
    float* shift1 = small + 384;
    float* scale2 = small + 512;
    float* shift2 = small + 640;
    float* d_num  = small + 768;      // num,den,ss0,ss1,ss2 : 5*64 contiguous
    float* d_den  = d_num + NB;
    float* d_ss0  = d_num + 2 * NB;
    float* d_ss1  = d_num + 3 * NB;
    float* d_ss2  = d_num + 4 * NB;
    float* d_s    = d_num + 5 * NB;   // S[NN x 2]
    float* d_pool = d_s + (size_t)NN * 2;  // out [NB x 256]

    const int NH = NN * HD;           // 4,194,304

    // weights -> transposed f16 B matrices
    k_prep_wt<<<HD, 256, 0, stream>>>(W1_rel, W1_root, Wt1);
    k_prep_wt<<<HD, 256, 0, stream>>>(W2_rel, W2_root, Wt2);

    // x0 = nf @ W_pre + b_pre
    k_pre<<<NH / 256, 256, 0, stream>>>(node_feature, W_pre, b_pre, buf0);

    // ---- layer 1 ----
    k_zero<<<NH / 256, 256, 0, stream>>>(buf1, NH);
    k_scatter<<<NE / 8, 256, 0, stream>>>(src, dst, edge_feature, buf0,
                                          nullptr, nullptr, buf1);
    k_pack<<<NH / 256, 256, 0, stream>>>(buf1, buf0, nullptr, nullptr, Ah);
    k_gemm<<<NN / 16, 256, 0, stream>>>(Ah, Wt1, b1_rel, buf1);   // y1 over agg1
    k_zero<<<1, 256, 0, stream>>>(d_sum, 256);
    k_stats<<<128, 256, 0, stream>>>(buf1, d_sum, d_sq);
    k_bn_finalize<<<1, 128, 0, stream>>>(d_sum, d_sq, g1, be1, scale1, shift1);

    // ---- layer 2 (BN+ReLU of y1 folded into gather/pack) ----
    k_zero<<<NH / 256, 256, 0, stream>>>(buf0, NH);
    k_scatter<<<NE / 8, 256, 0, stream>>>(src, dst, edge_feature, buf1,
                                          scale1, shift1, buf0);
    k_pack<<<NH / 256, 256, 0, stream>>>(buf0, buf1, scale1, shift1, Ah);
    k_gemm<<<NN / 16, 256, 0, stream>>>(Ah, Wt2, b2_rel, buf2);   // y2
    k_zero<<<1, 256, 0, stream>>>(d_sum, 256);
    k_stats<<<128, 256, 0, stream>>>(buf2, d_sum, d_sq);
    k_bn_finalize<<<1, 128, 0, stream>>>(d_sum, d_sq, g2, be2, scale2, shift2);

    // ---- mincut pooling (dense adjacency eliminated algebraically) ----
    k_zero<<<2, 256, 0, stream>>>(d_num, 5 * NB);
    k_pool<<<NN, 128, 0, stream>>>(buf2, scale2, shift2, W_pool, b_pool,
                                   d_s, d_ss0, d_ss1, d_ss2);
    k_out<<<NB, 256, 0, stream>>>(d_s, buf2, scale2, shift2, d_pool);
    k_edge_loss<<<NE / 256, 256, 0, stream>>>(src, dst, edge_feature, d_s,
                                              d_num, d_den);
    k_final<<<1, 256, 0, stream>>>(d_pool, W_post, b_post, d_num, d_den,
                                   d_ss0, d_ss1, d_ss2, dout);

    (void)in_sizes; (void)n_in; (void)out_size; (void)ws_size;
}
